// BoostEnhancedAttention_33603824124541
// MI455X (gfx1250) — matrix operand order repaired and checked by reading, not compiled, not run on hardware
//
#include <hip/hip_runtime.h>
#include <hip/hip_bf16.h>

// MI455X / gfx1250, wave32, WMMA bf16 path.
// Pipeline: (1) transpose+cvt values->Vt(bf16), (2) fused exp + P@values with
// deferred softmax normalization, (3) @W_v^T+b_v per head, (4) @W_o^T+b_o.
// b_v folds through softmax exactly (rows of P sum to 1 after normalization).

typedef __attribute__((ext_vector_type(16))) __bf16 v16bf;
typedef __attribute__((ext_vector_type(8)))  float  v8f;

union Frag {
  v16bf v;
  uint4 q[2];
  unsigned int u[8];
};

__device__ __forceinline__ unsigned short f32_bf16(float f) {
  unsigned int u = __float_as_uint(f);
  u += 0x7FFFu + ((u >> 16) & 1u);   // round to nearest even
  return (unsigned short)(u >> 16);
}
__device__ __forceinline__ unsigned int pack_bf16x2(float a, float b) {
  return (unsigned int)f32_bf16(a) | ((unsigned int)f32_bf16(b) << 16);
}

// ---------------------------------------------------------------------------
// Kernel 1: values [B][4096][512] f32  ->  Vt [B][512][4096] bf16 (transposed)
// 64x64 tiles through LDS; coalesced f32 reads, coalesced bf16 writes.
// ---------------------------------------------------------------------------
__global__ __launch_bounds__(256) void k_cvtT(const float* __restrict__ values,
                                              unsigned short* __restrict__ Vt) {
  __shared__ float tile[64][65];
  int bx  = blockIdx.x;
  int b   = bx >> 9;            // 512 tiles per batch (64 k-tiles * 8 n-tiles)
  int rem = bx & 511;
  int kt  = rem >> 3, nt = rem & 7;
  int k0 = kt * 64, n0 = nt * 64;
  int t = threadIdx.x;
  int r = t >> 2, cg = t & 3;

  const float* src = values + ((size_t)b * 4096 + (size_t)(k0 + r)) * 512 + n0;
#pragma unroll
  for (int i = 0; i < 4; ++i) {
    int c = cg * 16 + i * 4;
    float4 v = *(const float4*)(src + c);
    tile[r][c + 0] = v.x; tile[r][c + 1] = v.y;
    tile[r][c + 2] = v.z; tile[r][c + 3] = v.w;
  }
  __syncthreads();
  unsigned short* dst = Vt + ((size_t)b * 512 + (size_t)(n0 + r)) * 4096 + k0;
#pragma unroll
  for (int i = 0; i < 4; ++i) {
    int c = cg * 16 + i * 4;
    uint2 o;
    o.x = pack_bf16x2(tile[c + 0][r], tile[c + 1][r]);
    o.y = pack_bf16x2(tile[c + 2][r], tile[c + 3][r]);
    *(uint2*)(dst + c) = o;
  }
}

// ---------------------------------------------------------------------------
// Kernel 2: one block per (b,h), 512 threads = 16 waves.
// Single-pass: P_chunk = exp(att12*att3) (unnormalized) built in LDS; row sums
// accumulated via deterministic LDS reduction; normalization applied to the
// f32 WMMA accumulators in the epilogue. Each wave: 16 rows x 128 cols
// (8 tiles, 64 acc VGPRs) -> all 8 B fragments batched per K-step.
// Output O1 [b*h][64][512] bf16.
// ---------------------------------------------------------------------------
__global__ __launch_bounds__(512) void k_attn(const float* __restrict__ att12,
                                              const float* __restrict__ att3,
                                              const unsigned short* __restrict__ Vt,
                                              unsigned short* __restrict__ O1) {
  __shared__ float a12k[4096];                           // key-ordered att12 slice
  __shared__ __align__(16) unsigned short P[64][264];    // padded: no bank conflicts
  __shared__ float ssum[64];                             // running row sums
  __shared__ float red[64][8];                           // per-chunk partials

  int bh = blockIdx.x;
  int b = bh >> 3, h = bh & 7;
  int t = threadIdx.x;

  // att12[b,ch,cw,h,fh,fw] -> a12k[k], k = ch*256 + fh*64 + cw*4 + fw
  const float* a12g = att12 + (size_t)b * 32768;
#pragma unroll
  for (int i = 0; i < 8; ++i) {
    int k = t * 8 + i;
    int ch = k >> 8, fh = (k >> 6) & 3, cw = (k >> 2) & 15, fw = k & 3;
    a12k[k] = a12g[((ch * 16 + cw) * 8 + h) * 16 + fh * 4 + fw];
  }
  if (t < 64) ssum[t] = 0.f;
  __syncthreads();

  const float* a3g = att3 + ((size_t)b * 8 + h) * (64 * 256);

  int w = t >> 5, lane = t & 31;
  int wm = w & 3, wn = w >> 2;       // wave: rows [wm*16,+16), cols [wn*128,+128)
  int lrow = lane & 15, lhi = lane >> 4;

  v8f acc[8];
  for (int nt = 0; nt < 8; ++nt)
    for (int r = 0; r < 8; ++r) acc[nt][r] = 0.f;

  int qrow = t >> 3, sub = t & 7;                // P-build mapping: 8 thr/row
  const float* a3row = a3g + qrow * 256;

  for (int chunk = 0; chunk < 16; ++chunk) {     // K in chunks of 256 (ch fixed)
    // Build unnormalized P chunk (exp in [1,e)), accumulate row-sum partials.
    float psum = 0.f;
#pragma unroll
    for (int ii = 0; ii < 16; ++ii) {
      int kk = sub * 32 + ii * 2;                // kk,kk+1 share chcw
      float a3v = a3row[chunk * 16 + ((kk >> 2) & 15)];
      int k = chunk * 256 + kk;
      float x0 = __expf(a12k[k] * a3v);
      float x1 = __expf(a12k[k + 1] * a3v);
      psum += x0 + x1;
      *(unsigned int*)&P[qrow][kk] = pack_bf16x2(x0, x1);
    }
    red[qrow][sub] = psum;
    __syncthreads();
    if (t < 64) {                                // deterministic tree-free merge
      float s = 0.f;
#pragma unroll
      for (int j = 0; j < 8; ++j) s += red[t][j];
      ssum[t] += s;
    }

    const unsigned short* vb = Vt + (size_t)b * 512 * 4096 + chunk * 256;
#pragma unroll
    for (int step = 0; step < 8; ++step) {       // K substeps of 32
      Frag A;
      int ka = step * 32 + lhi * 16;
      A.q[0] = *(const uint4*)&P[wm * 16 + lrow][ka];
      A.q[1] = *(const uint4*)&P[wm * 16 + lrow][ka + 8];
      Frag Bf[8];                                // batch all B frags: one wait
#pragma unroll
      for (int nt = 0; nt < 8; ++nt) {
        int n = wn * 128 + nt * 16 + lrow;
        const unsigned short* vp = vb + (size_t)n * 4096 + step * 32 + lhi * 16;
        Bf[nt].q[0] = *(const uint4*)vp;
        Bf[nt].q[1] = *(const uint4*)(vp + 8);
      }
#pragma unroll
      for (int nt = 0; nt < 8; ++nt)
        acc[nt] = __builtin_amdgcn_wmma_f32_16x16x32_bf16(
            false, A.v, false, Bf[nt].v, (short)0, acc[nt], false, false);
    }
    __syncthreads();                             // P + red reuse guard
  }

  // Deferred softmax normalization on accumulators, then store bf16.
  float rs[8];
#pragma unroll
  for (int r = 0; r < 8; ++r) rs[r] = 1.0f / ssum[wm * 16 + lhi * 8 + r];
  unsigned short* o1 = O1 + (size_t)bh * 64 * 512;
  for (int nt = 0; nt < 8; ++nt) {
    int col = wn * 128 + nt * 16 + lrow;
#pragma unroll
    for (int r = 0; r < 8; ++r) {
      int row = wm * 16 + r + lhi * 8;
      o1[(size_t)row * 512 + col] = f32_bf16(acc[nt][r] * rs[r]);
    }
  }
}

// ---------------------------------------------------------------------------
// Kernel 3: per (b,h): U[b,q,h*64+dv] = O1[bh] @ W_v_h^T + b_v_h  (64x512 @ 512x64)
// ---------------------------------------------------------------------------
__global__ __launch_bounds__(256) void k_headmix(const unsigned short* __restrict__ O1,
                                                 const float* __restrict__ Wv,
                                                 const float* __restrict__ bv,
                                                 unsigned short* __restrict__ U) {
  int bh = blockIdx.x;
  int b = bh >> 3, h = bh & 7;
  int t = threadIdx.x;
  int w = t >> 5, lane = t & 31;
  int mtile = w & 3, npair = w >> 2;   // 4 M-tiles x (2 waves x 2 N-tiles)
  int lrow = lane & 15, lhi = lane >> 4;

  v8f acc[2];
  for (int ni = 0; ni < 2; ++ni)
    for (int r = 0; r < 8; ++r) acc[ni][r] = 0.f;

  const unsigned short* a =
      O1 + (size_t)bh * 64 * 512 + (size_t)(mtile * 16 + lrow) * 512;
#pragma unroll
  for (int step = 0; step < 16; ++step) {
    Frag A;
    int ka = step * 32 + lhi * 16;
    A.q[0] = *(const uint4*)(a + ka);
    A.q[1] = *(const uint4*)(a + ka + 8);
    Frag Bf[2];                        // both B frags loaded before the WMMAs
#pragma unroll
    for (int ni = 0; ni < 2; ++ni) {
      int hd = h * 64 + (npair * 2 + ni) * 16 + lrow;   // row of W_v
      const float* wp = Wv + (size_t)hd * 512 + step * 32 + lhi * 16;
#pragma unroll
      for (int jj = 0; jj < 4; ++jj) {
        float4 f = *(const float4*)(wp + jj * 4);
        Bf[ni].u[jj * 2 + 0] = pack_bf16x2(f.x, f.y);
        Bf[ni].u[jj * 2 + 1] = pack_bf16x2(f.z, f.w);
      }
    }
#pragma unroll
    for (int ni = 0; ni < 2; ++ni)
      acc[ni] = __builtin_amdgcn_wmma_f32_16x16x32_bf16(
          false, A.v, false, Bf[ni].v, (short)0, acc[ni], false, false);
  }

  unsigned short* ub = U + (size_t)b * 64 * 512;
  for (int ni = 0; ni < 2; ++ni) {
    int col = (npair * 2 + ni) * 16 + lrow;
    int hd = h * 64 + col;
    float bvv = bv[hd];
#pragma unroll
    for (int r = 0; r < 8; ++r) {
      int row = mtile * 16 + r + lhi * 8;
      ub[(size_t)row * 512 + hd] = f32_bf16(acc[ni][r] + bvv);
    }
  }
}

// ---------------------------------------------------------------------------
// Kernel 4: per batch: out[b] = U[b] @ W_o^T + b_o   (64x512 @ 512x512, f32 out)
// 512 threads = 16 waves, 8 tiles/wave, batched B fragments per K-step.
// ---------------------------------------------------------------------------
__global__ __launch_bounds__(512) void k_out(const unsigned short* __restrict__ U,
                                             const float* __restrict__ Wo,
                                             const float* __restrict__ bo,
                                             float* __restrict__ out) {
  int b = blockIdx.x;
  int t = threadIdx.x;
  int w = t >> 5, lane = t & 31;
  int wm = w & 3, wn = w >> 2;
  int lrow = lane & 15, lhi = lane >> 4;

  v8f acc[8];
  for (int nt = 0; nt < 8; ++nt)
    for (int r = 0; r < 8; ++r) acc[nt][r] = 0.f;

  const unsigned short* ub =
      U + (size_t)b * 64 * 512 + (size_t)(wm * 16 + lrow) * 512;
#pragma unroll
  for (int step = 0; step < 16; ++step) {
    Frag A;
    int ka = step * 32 + lhi * 16;
    A.q[0] = *(const uint4*)(ub + ka);
    A.q[1] = *(const uint4*)(ub + ka + 8);
    Frag Bf[8];
#pragma unroll
    for (int nt = 0; nt < 8; ++nt) {
      int dm = wn * 128 + nt * 16 + lrow;               // row of W_o
      const float* wp = Wo + (size_t)dm * 512 + step * 32 + lhi * 16;
#pragma unroll
      for (int jj = 0; jj < 4; ++jj) {
        float4 f = *(const float4*)(wp + jj * 4);
        Bf[nt].u[jj * 2 + 0] = pack_bf16x2(f.x, f.y);
        Bf[nt].u[jj * 2 + 1] = pack_bf16x2(f.z, f.w);
      }
    }
#pragma unroll
    for (int nt = 0; nt < 8; ++nt)
      acc[nt] = __builtin_amdgcn_wmma_f32_16x16x32_bf16(
          false, A.v, false, Bf[nt].v, (short)0, acc[nt], false, false);
  }

  float* ob = out + (size_t)b * 64 * 512;
  for (int nt = 0; nt < 8; ++nt) {
    int dm = wn * 128 + nt * 16 + lrow;
    float bov = bo[dm];
#pragma unroll
    for (int r = 0; r < 8; ++r) {
      int row = wm * 16 + r + lhi * 8;
      ob[(size_t)row * 512 + dm] = acc[nt][r] + bov;
    }
  }
}

// ---------------------------------------------------------------------------
extern "C" void kernel_launch(void* const* d_in, const int* in_sizes, int n_in,
                              void* d_out, int out_size, void* d_ws, size_t ws_size,
                              hipStream_t stream) {
  const float* att12  = (const float*)d_in[0];
  const float* att3   = (const float*)d_in[1];
  const float* values = (const float*)d_in[2];
  const float* Wv     = (const float*)d_in[3];
  const float* bv     = (const float*)d_in[4];
  const float* Wo     = (const float*)d_in[5];
  const float* bo     = (const float*)d_in[6];
  float* out = (float*)d_out;

  char* ws = (char*)d_ws;
  unsigned short* Vt = (unsigned short*)ws;                              // 128 MiB
  unsigned short* O1 = (unsigned short*)(ws + (size_t)134217728);        //  16 MiB
  unsigned short* U  = (unsigned short*)(ws + (size_t)134217728 + 16777216); // 2 MiB

  k_cvtT   <<<16384, 256, 0, stream>>>(values, Vt);
  k_attn   <<<256,   512, 0, stream>>>(att12, att3, Vt, O1);
  k_headmix<<<256,   256, 0, stream>>>(O1, Wv, bv, U);
  k_out    <<<32,    512, 0, stream>>>(U, Wo, bo, out);
}
